// QuantumEncoderLayer_81939386073593
// MI455X (gfx1250) — matrix operand order repaired and checked by reading, not compiled, not run on hardware
//
#include <hip/hip_runtime.h>
#include <math.h>

// ---------------------------------------------------------------------------
// CDNA5 (gfx1250) types for WMMA bf16 16x16x32, f32 accumulate
// ---------------------------------------------------------------------------
typedef __bf16 bf16_t;
typedef __attribute__((ext_vector_type(16))) bf16_t v16bf;
typedef __attribute__((ext_vector_type(8)))  bf16_t v8bf;
typedef __attribute__((ext_vector_type(8)))  float  v8f;

__device__ __forceinline__ v8f wmma_bf16(v16bf a, v16bf b, v8f c) {
  // (neg_a, A, neg_b, B, c_mod, C, reuse_a, reuse_b)
  return __builtin_amdgcn_wmma_f32_16x16x32_bf16(false, a, false, b,
                                                 (short)0, c, false, false);
}

__device__ __forceinline__ void zero8(v8f& a) {
#pragma unroll
  for (int i = 0; i < 8; ++i) a[i] = 0.0f;
}

__device__ __forceinline__ v16bf cat8(v8bf lo, v8bf hi) {
  return __builtin_shufflevector(lo, hi, 0, 1, 2, 3, 4, 5, 6, 7,
                                 8, 9, 10, 11, 12, 13, 14, 15);
}

// A fragment (16x32 bf16) from a bf16 row in global memory.
// elems 0..15 = K[k0+half*16 ..]
__device__ __forceinline__ v16bf load_a_bf_g(const bf16_t* __restrict__ arow,
                                             int k0, int half) {
  return *(const v16bf*)(arow + k0 + half * 16);
}

// A fragment from an LDS bf16 row (same contiguous layout).
__device__ __forceinline__ void load_a_frag_lds(v16bf& a, const bf16_t* rowp,
                                                int k0, int half) {
#pragma unroll
  for (int i = 0; i < 16; ++i) a[i] = rowp[k0 + half * 16 + i];
}

__device__ __forceinline__ float gelu_exact(float x) {
  return 0.5f * x * (1.0f + erff(x * 0.70710678118654752f));
}

// ---------------------------------------------------------------------------
// fp32 -> bf16 bulk conversion (8 elements / thread, 128-bit IO)
// ---------------------------------------------------------------------------
__global__ void cvt_bf16_kernel(const float* __restrict__ src,
                                bf16_t* __restrict__ dst, long n8) {
  const long t = (long)blockIdx.x * 256 + threadIdx.x;
  if (t >= n8) return;
  const float4* s = (const float4*)(src + t * 8);
  const float4 a = s[0], b = s[1];
  v8bf o;
  o[0] = (bf16_t)a.x; o[1] = (bf16_t)a.y; o[2] = (bf16_t)a.z; o[3] = (bf16_t)a.w;
  o[4] = (bf16_t)b.x; o[5] = (bf16_t)b.y; o[6] = (bf16_t)b.z; o[7] = (bf16_t)b.w;
  *(v8bf*)(dst + t * 8) = o;
}

// ---------------------------------------------------------------------------
// GEMM: C[M,N] = act(A[M,K] @ W[N,K]^T + bias[N]) (+ residual[M,N])
// A, W are bf16 (pre-converted); C is fp32 or bf16 per OUT_BF16.
// Block = 256 threads (8 waves): 64(M) x 128(N); wave = 16x64 (4 accs).
// M%64==0, N%128==0, K%32==0 for all uses here.
// ---------------------------------------------------------------------------
template <int ACT, bool HAS_RES, bool OUT_BF16>
__global__ void gemm_wmma_kernel(const bf16_t* __restrict__ A,
                                 const bf16_t* __restrict__ W,
                                 const float* __restrict__ bias,
                                 const float* __restrict__ res,
                                 void* __restrict__ Cv,
                                 int M, int N, int K) {
  const int lane = threadIdx.x & 31;
  const int wave = threadIdx.x >> 5;
  const int wm = wave & 3;   // 4 waves along M
  const int wn = wave >> 2;  // 2 waves along N
  const int m0 = blockIdx.y * 64 + wm * 16;
  const int n0 = blockIdx.x * 128 + wn * 64;
  const int am = lane & 15;
  const int half = lane >> 4;

  const bf16_t* arow = A + (size_t)(m0 + am) * K;

  v8f acc[4];
#pragma unroll
  for (int t = 0; t < 4; ++t) zero8(acc[t]);

#pragma unroll 2
  for (int k0 = 0; k0 < K; k0 += 32) {
    if (k0 + 32 < K) __builtin_prefetch(arow + k0 + 32, 0, 1);  // global_prefetch
    const v16bf a = load_a_bf_g(arow, k0, half);
#pragma unroll
    for (int t = 0; t < 4; ++t) {
      const v16bf b = *(const v16bf*)(W + (size_t)(n0 + t * 16 + am) * K
                                        + k0 + half * 16);
      acc[t] = wmma_bf16(a, b, acc[t]);
    }
  }

#pragma unroll
  for (int t = 0; t < 4; ++t) {
    const int col = n0 + t * 16 + am;
    const float bv = bias[col];
#pragma unroll
    for (int r = 0; r < 8; ++r) {
      const int row = m0 + r + half * 8;
      float v = acc[t][r] + bv;
      if (ACT == 1) v = gelu_exact(v);
      if (HAS_RES) v += res[(size_t)row * N + col];
      if (OUT_BF16) ((bf16_t*)Cv)[(size_t)row * N + col] = (bf16_t)v;
      else          ((float*)Cv)[(size_t)row * N + col] = v;
    }
  }
}

// ---------------------------------------------------------------------------
// Attention over bf16 qkv [B*S,1536] = [q|k|v], 4 heads x 128.
// One block per (16-query tile, b*4+h). Scores (fp32) stay in LDS; output bf16.
// ---------------------------------------------------------------------------
__global__ void attn_kernel(const bf16_t* __restrict__ qkv,
                            bf16_t* __restrict__ out) {
  __shared__ float  s_sc[16][516];
  __shared__ bf16_t s_q[16][132];
  __shared__ bf16_t s_p[16][520];

  const int lane = threadIdx.x & 31;
  const int wave = threadIdx.x >> 5;
  const int q0 = blockIdx.x * 16;
  const int bh = blockIdx.y;
  const int b = bh >> 2, h = bh & 3;
  const float scale = 0.088388347648318447f;  // 1/sqrt(128)

  // stage pre-scaled Q into LDS
  for (int t = threadIdx.x; t < 16 * 128; t += 256) {
    const int r = t >> 7, c = t & 127;
    const float qv = (float)qkv[(size_t)(b * 512 + q0 + r) * 1536 + h * 128 + c];
    s_q[r][c] = (bf16_t)(qv * scale);
  }
  __syncthreads();

  const int am = lane & 15;
  const int half = lane >> 4;

  // scores = Qs @ K^T : 32 n-tiles of 16 keys, 4 per wave
#pragma unroll
  for (int tt = 0; tt < 4; ++tt) {
    const int nt = wave * 4 + tt;
    v8f acc; zero8(acc);
#pragma unroll
    for (int k0 = 0; k0 < 128; k0 += 32) {
      v16bf a;
      load_a_frag_lds(a, &s_q[am][0], k0, half);
      const v16bf bf = *(const v16bf*)(qkv + (size_t)(b * 512 + nt * 16 + am) * 1536
                                           + 512 + h * 128 + k0 + half * 16);
      acc = wmma_bf16(a, bf, acc);
    }
#pragma unroll
    for (int r = 0; r < 8; ++r) s_sc[r + half * 8][nt * 16 + am] = acc[r];
  }
  __syncthreads();

  // softmax: 2 rows per wave, lane-parallel over 512 cols
#pragma unroll
  for (int rr = 0; rr < 2; ++rr) {
    const int r = wave * 2 + rr;
    float vals[16];
    float mx = -3.4e38f;
#pragma unroll
    for (int i = 0; i < 16; ++i) {
      vals[i] = s_sc[r][lane + i * 32];
      mx = fmaxf(mx, vals[i]);
    }
#pragma unroll
    for (int off = 16; off; off >>= 1) mx = fmaxf(mx, __shfl_xor(mx, off, 32));
    float sum = 0.0f;
#pragma unroll
    for (int i = 0; i < 16; ++i) { vals[i] = __expf(vals[i] - mx); sum += vals[i]; }
#pragma unroll
    for (int off = 16; off; off >>= 1) sum += __shfl_xor(sum, off, 32);
    const float inv = 1.0f / sum;
#pragma unroll
    for (int i = 0; i < 16; ++i) s_p[r][lane + i * 32] = (bf16_t)(vals[i] * inv);
  }
  __syncthreads();

  // O = P @ V : 8 dh-tiles of 16, one per wave
  v8f acc; zero8(acc);
  for (int k0 = 0; k0 < 512; k0 += 32) {
    v16bf a;
    load_a_frag_lds(a, &s_p[am][0], k0, half);
    v16bf bf;
#pragma unroll
    for (int i = 0; i < 16; ++i) {
      const int s = k0 + half * 16 + i;
      bf[i] = qkv[(size_t)(b * 512 + s) * 1536 + 1024 + h * 128 + wave * 16 + am];
    }
    acc = wmma_bf16(a, bf, acc);
  }
#pragma unroll
  for (int r = 0; r < 8; ++r) {
    out[(size_t)(b * 512 + q0 + r + half * 8) * 512 + h * 128 + wave * 16 + am] =
        (bf16_t)acc[r];
  }
}

// ---------------------------------------------------------------------------
// Row layernorm over d=512: one wave per row, 8 rows per block.
// ---------------------------------------------------------------------------
__global__ void ln_kernel(const float* __restrict__ x, const float* __restrict__ g,
                          const float* __restrict__ bb, float* __restrict__ y) {
  const int lane = threadIdx.x & 31;
  const int wave = threadIdx.x >> 5;
  const size_t row = (size_t)blockIdx.x * 8 + wave;
  const float* xr = x + row * 512;
  float v[16];
  float s = 0.0f, s2 = 0.0f;
#pragma unroll
  for (int i = 0; i < 16; ++i) {
    const float t = xr[lane + i * 32];
    v[i] = t; s += t; s2 += t * t;
  }
#pragma unroll
  for (int off = 16; off; off >>= 1) {
    s  += __shfl_xor(s, off, 32);
    s2 += __shfl_xor(s2, off, 32);
  }
  const float mean = s * (1.0f / 512.0f);
  const float var  = s2 * (1.0f / 512.0f) - mean * mean;
  const float inv  = rsqrtf(var + 1e-5f);
#pragma unroll
  for (int i = 0; i < 16; ++i) {
    const int c = lane + i * 32;
    y[row * 512 + c] = (v[i] - mean) * inv * g[c] + bb[c];
  }
}

// ---------------------------------------------------------------------------
// 8-qubit statevector sim: one wave per row, 8 complex amps per lane.
// idx = (lane<<3)|j. Gates on bits>=3 use shfl_xor; bits<3 in-register.
// ---------------------------------------------------------------------------
__device__ __forceinline__ void apply_rx(float (&re)[8], float (&im)[8],
                                         float th, int wire, int lane) {
  float s, c;
  __sincosf(th * 0.5f, &s, &c);
  const int b = 7 - wire;
  if (b >= 3) {
    const int m = 1 << (b - 3);
#pragma unroll
    for (int j = 0; j < 8; ++j) {
      const float pre = __shfl_xor(re[j], m, 32);
      const float pim = __shfl_xor(im[j], m, 32);
      const float nr = c * re[j] + s * pim;
      const float ni = c * im[j] - s * pre;
      re[j] = nr; im[j] = ni;
    }
  } else {
    const int m = 1 << b;
#pragma unroll
    for (int j = 0; j < 8; ++j) {
      if (!(j & m)) {
        const int j1 = j | m;
        const float r0 = re[j], i0 = im[j], r1 = re[j1], i1 = im[j1];
        re[j]  = c * r0 + s * i1;  im[j]  = c * i0 - s * r1;
        re[j1] = c * r1 + s * i0;  im[j1] = c * i1 - s * r0;
      }
    }
  }
  (void)lane;
}

__device__ __forceinline__ void apply_rz(float (&re)[8], float (&im)[8],
                                         float th, int wire, int lane) {
  float s, c;
  __sincosf(th * 0.5f, &s, &c);
  const int b = 7 - wire;
#pragma unroll
  for (int j = 0; j < 8; ++j) {
    const int idx = (lane << 3) | j;
    const float t = ((idx >> b) & 1) ? -1.0f : 1.0f;
    const float nr = c * re[j] + t * s * im[j];
    const float ni = c * im[j] - t * s * re[j];
    re[j] = nr; im[j] = ni;
  }
}

__device__ __forceinline__ void apply_cnot(float (&re)[8], float (&im)[8],
                                           int ctrl, int lane) {
  const int cb = 7 - ctrl;
  const int tb = 6 - ctrl;
  if (tb >= 3) {
    const int m = 1 << (tb - 3);
#pragma unroll
    for (int j = 0; j < 8; ++j) {
      const float pre = __shfl_xor(re[j], m, 32);
      const float pim = __shfl_xor(im[j], m, 32);
      const int idx = (lane << 3) | j;
      const bool cset = ((idx >> cb) & 1) != 0;
      re[j] = cset ? pre : re[j];
      im[j] = cset ? pim : im[j];
    }
  } else {
    const int m = 1 << tb;
#pragma unroll
    for (int j = 0; j < 8; ++j) {
      if (!(j & m)) {
        const int j1 = j | m;
        const int idx = (lane << 3) | j;
        const bool cset = ((idx >> cb) & 1) != 0;
        const float r0 = re[j], i0 = im[j], r1 = re[j1], i1 = im[j1];
        re[j]  = cset ? r1 : r0;  im[j]  = cset ? i1 : i0;
        re[j1] = cset ? r0 : r1;  im[j1] = cset ? i0 : i1;
      }
    }
  }
}

__global__ void quantum_kernel(const float* __restrict__ xn,
                               const float* __restrict__ qin_w,
                               const float* __restrict__ qin_b,
                               const float* __restrict__ qw,
                               float* __restrict__ z) {
  const int lane = threadIdx.x & 31;
  const int wave = threadIdx.x >> 5;
  const size_t row = (size_t)blockIdx.x * 8 + wave;

  // angles = xn[row] @ qin_w^T + qin_b
  const float* xr = xn + row * 512;
  float ang[8];
#pragma unroll
  for (int q = 0; q < 8; ++q) {
    float p = 0.0f;
    for (int c = lane; c < 512; c += 32) p += xr[c] * qin_w[q * 512 + c];
#pragma unroll
    for (int off = 16; off; off >>= 1) p += __shfl_xor(p, off, 32);
    ang[q] = p + qin_b[q];
  }

  float re[8], im[8];
#pragma unroll
  for (int j = 0; j < 8; ++j) { re[j] = 0.0f; im[j] = 0.0f; }
  if (lane == 0) re[0] = 1.0f;

#pragma unroll
  for (int i = 0; i < 8; ++i) {
    apply_rx(re, im, ang[i], i, lane);
    apply_rz(re, im, ang[i], i, lane);
  }
#pragma unroll
  for (int l = 0; l < 4; ++l) {
#pragma unroll
    for (int i = 0; i < 8; ++i) {
      apply_rx(re, im, qw[l * 16 + i], i, lane);
      apply_rz(re, im, qw[l * 16 + 8 + i], i, lane);
    }
#pragma unroll
    for (int i = 0; i < 7; ++i) apply_cnot(re, im, i, lane);
  }

  float zz[8];
#pragma unroll
  for (int i = 0; i < 8; ++i) zz[i] = 0.0f;
#pragma unroll
  for (int j = 0; j < 8; ++j) {
    const float pr = re[j] * re[j] + im[j] * im[j];
    const int idx = (lane << 3) | j;
#pragma unroll
    for (int i = 0; i < 8; ++i) zz[i] += (((idx >> (7 - i)) & 1) ? -pr : pr);
  }
#pragma unroll
  for (int i = 0; i < 8; ++i) {
#pragma unroll
    for (int off = 16; off; off >>= 1) zz[i] += __shfl_xor(zz[i], off, 32);
    if (lane == 0) z[row * 8 + i] = zz[i];
  }
}

// x2 = x1 + z @ qout_w^T + qout_b; dual-write fp32 (residual) + bf16 (GEMM A)
__global__ void qproj_kernel(const float* __restrict__ x1, const float* __restrict__ z,
                             const float* __restrict__ qow, const float* __restrict__ qob,
                             float* __restrict__ x2, bf16_t* __restrict__ x2b) {
  const size_t t = (size_t)blockIdx.x * 256 + threadIdx.x;
  const size_t row = t >> 9;
  const int col = (int)(t & 511);
  float acc = qob[col];
  const float* zr = z + row * 8;
  const float* wr = qow + (size_t)col * 8;
#pragma unroll
  for (int i = 0; i < 8; ++i) acc += zr[i] * wr[i];
  const float v = x1[t] + acc;
  x2[t] = v;
  x2b[t] = (bf16_t)v;
}

// ---------------------------------------------------------------------------
// Launch: full forward pass.
// ---------------------------------------------------------------------------
extern "C" void kernel_launch(void* const* d_in, const int* in_sizes, int n_in,
                              void* d_out, int out_size, void* d_ws, size_t ws_size,
                              hipStream_t stream) {
  const float* x          = (const float*)d_in[0];
  const float* attn_in_w  = (const float*)d_in[1];
  const float* attn_in_b  = (const float*)d_in[2];
  const float* attn_out_w = (const float*)d_in[3];
  const float* attn_out_b = (const float*)d_in[4];
  const float* ln1_g      = (const float*)d_in[5];
  const float* ln1_b      = (const float*)d_in[6];
  const float* ln2_g      = (const float*)d_in[7];
  const float* ln2_b      = (const float*)d_in[8];
  const float* ln3_g      = (const float*)d_in[9];
  const float* ln3_b      = (const float*)d_in[10];
  const float* qin_w      = (const float*)d_in[11];
  const float* qin_b      = (const float*)d_in[12];
  const float* qweights   = (const float*)d_in[13];
  const float* qout_w     = (const float*)d_in[14];
  const float* qout_b     = (const float*)d_in[15];
  const float* ffn_w1     = (const float*)d_in[16];
  const float* ffn_b1     = (const float*)d_in[17];
  const float* ffn_w2     = (const float*)d_in[18];
  const float* ffn_b2     = (const float*)d_in[19];
  float* out = (float*)d_out;

  const int M = 32 * 512;  // 16384 rows
  const size_t ROWS = (size_t)M;

  // ---- workspace layout -------------------------------------------------
  // fp32 region
  float* f_t0 = (float*)d_ws;                 // t0, later reused as xn
  float* f_x1 = f_t0 + ROWS * 512;            // x1, later reused as t1
  float* f_x2 = f_x1 + ROWS * 512;            // x2 (fp32, FFN2 residual)
  float* f_z  = f_x2 + ROWS * 512;            // quantum expvals [16384,8]
  // bf16 region
  bf16_t* b_w_in  = (bf16_t*)(f_z + ROWS * 8);      // 1536*512
  bf16_t* b_w_out = b_w_in  + (size_t)1536 * 512;   // 512*512
  bf16_t* b_w1    = b_w_out + (size_t)512 * 512;    // 2048*512
  bf16_t* b_w2    = b_w1    + (size_t)2048 * 512;   // 512*2048
  bf16_t* b_act   = b_w2    + (size_t)512 * 2048;   // ROWS*512: x -> attn -> x2
  bf16_t* b_qkv   = b_act   + ROWS * 512;           // ROWS*1536
  bf16_t* b_h     = b_qkv   + ROWS * 1536;          // ROWS*2048
  float* f_xn = f_t0;   // reuse (t0 dead after LN1)
  float* f_t1 = f_x1;   // reuse (x1 dead after qproj)

  dim3 blk(256);
  auto cvt = [&](const float* s, bf16_t* d, size_t n) {
    const long n8 = (long)(n / 8);
    cvt_bf16_kernel<<<dim3((unsigned)((n8 + 255) / 256)), blk, 0, stream>>>(s, d, n8);
  };

  // 0) one-time bf16 conversions (weights + x)
  cvt(attn_in_w,  b_w_in,  (size_t)1536 * 512);
  cvt(attn_out_w, b_w_out, (size_t)512 * 512);
  cvt(ffn_w1,     b_w1,    (size_t)2048 * 512);
  cvt(ffn_w2,     b_w2,    (size_t)512 * 2048);
  cvt(x,          b_act,   ROWS * 512);

  // 1) qkv(bf16) = x @ in_w^T + in_b
  gemm_wmma_kernel<0, false, true><<<dim3(1536 / 128, M / 64), blk, 0, stream>>>(
      b_act, b_w_in, attn_in_b, nullptr, b_qkv, M, 1536, 512);

  // 2) attention -> b_act (bf16; x bf16 dead after GEMM1)
  attn_kernel<<<dim3(512 / 16, 32 * 4), blk, 0, stream>>>(b_qkv, b_act);

  // 3) t0 = attn @ out_w^T + out_b + x (fp32)
  gemm_wmma_kernel<0, true, false><<<dim3(512 / 128, M / 64), blk, 0, stream>>>(
      b_act, b_w_out, attn_out_b, x, f_t0, M, 512, 512);

  // 4) x1 = LN1(t0); xn = LN3(x1)
  ln_kernel<<<dim3(M / 8), blk, 0, stream>>>(f_t0, ln1_g, ln1_b, f_x1);
  ln_kernel<<<dim3(M / 8), blk, 0, stream>>>(f_x1, ln3_g, ln3_b, f_xn);

  // 5) quantum expvals -> f_z
  quantum_kernel<<<dim3(M / 8), blk, 0, stream>>>(f_xn, qin_w, qin_b, qweights, f_z);

  // 6) x2 = x1 + z @ qout_w^T + qout_b  (fp32 + bf16 into b_act)
  qproj_kernel<<<dim3((unsigned)((ROWS * 512) / 256)), blk, 0, stream>>>(
      f_x1, f_z, qout_w, qout_b, f_x2, b_act);

  // 7) h(bf16) = gelu(x2 @ w1^T + b1)
  gemm_wmma_kernel<1, false, true><<<dim3(2048 / 128, M / 64), blk, 0, stream>>>(
      b_act, b_w1, ffn_b1, nullptr, b_h, M, 2048, 512);

  // 8) t1 = h @ w2^T + b2 + x2 (fp32)
  gemm_wmma_kernel<0, true, false><<<dim3(512 / 128, M / 64), blk, 0, stream>>>(
      b_h, b_w2, ffn_b2, f_x2, f_t1, M, 512, 2048);

  // 9) out = LN2(t1)
  ln_kernel<<<dim3(M / 8), blk, 0, stream>>>(f_t1, ln2_g, ln2_b, out);

  (void)in_sizes; (void)n_in; (void)out_size; (void)ws_size;
}
